// Net_76244259439077
// MI455X (gfx1250) — matrix operand order repaired and checked by reading, not compiled, not run on hardware
//
#include <hip/hip_runtime.h>

// ---------------------------------------------------------------------------
// CDNA5 (gfx1250) implementation of the SConvLSTM spiking net.
// Convs + FC1 run as implicit GEMM on v_wmma_f32_16x16x32_bf16 (f32 accum).
// LDS tiles are stored pre-swizzled into WMMA fragment order (all fragment
// reads are ds_load_b128); weight tiles are staged with the gfx1250 async
// global->LDS path (global_load_async_to_lds_b128 / s_wait_asynccnt).
// ---------------------------------------------------------------------------

typedef __attribute__((ext_vector_type(16))) __bf16        v16bf;
typedef __attribute__((ext_vector_type(8)))  float         v8f;
typedef __attribute__((ext_vector_type(4)))  unsigned int  v4u;
typedef __attribute__((ext_vector_type(4)))  float         v4f;

#define THR  0.05f
#define BETA 0.9181805491303656f

#define LDS_A_HALVES (64 * 2 * 16)            // 4 KB swizzled A tile
#define LDS_B_HALVES (2 * 32 * 16)            // 2 KB swizzled B tile
#define LDS_BYTES    ((LDS_A_HALVES + LDS_B_HALVES) * 2)

__device__ __forceinline__ float sigf(float x) { return 1.0f / (1.0f + __expf(-x)); }

// round-to-nearest-even f32 -> bf16 bit pattern
__device__ __forceinline__ unsigned short f2bf(float f) {
    union { float f; unsigned u; } uf; uf.f = f;
    unsigned r = uf.u + 0x7FFFu + ((uf.u >> 16) & 1u);
    return (unsigned short)(r >> 16);
}

union FragU { v4u u[2]; v16bf v; };
union Pack8 { unsigned short s[8]; v4u u; };

// 16B async copy global -> LDS (ASYNCcnt); ldsoff = byte offset in WG LDS.
__device__ __forceinline__ void async_g2l_b128(unsigned ldsoff, const void* gaddr) {
    asm volatile("global_load_async_to_lds_b128 %0, %1, off"
                 :: "v"(ldsoff), "v"((unsigned long long)gaddr) : "memory");
}
__device__ __forceinline__ void wait_async0() {
    asm volatile("s_wait_asynccnt 0x0" ::: "memory");
}

// ---------------------------------------------------------------------------
// Weight repack: [Cout][Ktot] f32 -> [Cout][Kpad] bf16 bits, zero padded.
// ---------------------------------------------------------------------------
__global__ void repack_bf16(const float* __restrict__ src, unsigned short* __restrict__ dst,
                            int Cout, int Ktot, int Kpad) {
    int idx = blockIdx.x * 256 + threadIdx.x;
    if (idx >= Cout * Kpad) return;
    int co = idx / Kpad, k = idx - co * Kpad;
    float v = (k < Ktot) ? src[co * Ktot + k] : 0.0f;
    dst[idx] = f2bf(v);
}

// Fragment read: 32 contiguous bytes per lane -> 2x ds_load_b128
__device__ __forceinline__ v16bf load_frag(const unsigned short* p) {
    FragU f;
    f.u[0] = *(const v4u*)(p);
    f.u[1] = *(const v4u*)(p + 8);
    return f.v;
}

// ---------------------------------------------------------------------------
// Implicit-GEMM conv producing the 4*Chid gate pre-activations (+bias).
//   gates[co][n] , n = (b*H + y)*W + x  (flat over B*H*W)
// Block = 128 threads = 4 waves. WG tile: M=64 x N=32 (2 N-tiles per wave).
// LDS layouts (dynamic LDS, base offset 0):
//   sA[(m*2+khalf)*16 + j] : K=khalf*16+j (contiguous)
//   sB[(ti*32+lane)*16 + j]: lane=(K>=16)*16+n ; j=K&15
// ---------------------------------------------------------------------------
__global__ __launch_bounds__(128)
void conv_gates_wmma(const float* __restrict__ in,    // [B][Cin ][H][W]
                     const float* __restrict__ hid,   // [B][Chid][H][W]
                     const unsigned short* __restrict__ Wp, // [Cout][Kpad] bf16
                     const float* __restrict__ bias,  // [Cout]
                     float* __restrict__ gates,       // [Cout][B*H*W]
                     int Bn, int Cin, int Chid, int H, int Wd, int Kpad) {
    extern __shared__ unsigned short sMem[];
    unsigned short* sA = sMem;                 // [64*2*16]
    unsigned short* sB = sMem + LDS_A_HALVES;  // [2*32*16]

    const int HW   = H * Wd;
    const int Npos = Bn * HW;
    const int Ccat = Cin + Chid;

    const int tid  = threadIdx.x;
    const int wave = tid >> 5;
    const int lane = tid & 31;

    const int Mblk  = blockIdx.y * 64;
    const int Nbase = blockIdx.x * 32;      // 32 positions, same image row (W%32==0)
    const int b   = Nbase / HW;
    const int rem = Nbase - b * HW;
    const int y   = rem / Wd;
    const int x0  = rem - y * Wd;

    // A staging decode (two 8-half groups per thread)
    const int g0  = tid,        g1  = tid + 128;
    const int am0 = g0 >> 2,    am1 = g1 >> 2;
    const int aq0 = g0 & 3,     aq1 = g1 & 3;
    const unsigned adst0 = (unsigned)((am0 * 2 + (aq0 >> 1)) * 16 + (aq0 & 1) * 8) * 2;
    const unsigned adst1 = (unsigned)((am1 * 2 + (aq1 >> 1)) * 16 + (aq1 & 1) * 8) * 2;
    const unsigned short* asrc0 = Wp + (size_t)(Mblk + am0) * Kpad + aq0 * 8;
    const unsigned short* asrc1 = Wp + (size_t)(Mblk + am1) * Kpad + aq1 * 8;

    // B staging decode: 8 consecutive K taps of one dest lane
    const int dflat = tid * 8;               // 1024 halves / 128 threads
    const int ti_st = dflat >> 9;
    const int remb  = dflat & 511;
    const int L_st  = remb >> 4;
    const int j0_st = remb & 15;
    const int nn_st = L_st & 15;
    const int kk0   = ((L_st >> 4) << 4) + j0_st;   // K offset within 32-block
    const int xg    = x0 + ti_st * 16 + nn_st;      // global x of this column

    // incremental (channel, tap) decode: kg = 32*iter + kk0 + jj
    int cc = kk0 / 9;
    int rr = kk0 - cc * 9;

    v8f acc0 = {}, acc1 = {};

    for (int k0 = 0; k0 < Kpad; k0 += 32) {
        // ---- stage A asynchronously: 2x 16B global->LDS per thread ----
        async_g2l_b128(adst0, asrc0 + k0);
        async_g2l_b128(adst1, asrc1 + k0);

        // ---- stage B: im2col, 8 taps/thread, one 16B ds store ----
        {
            Pack8 pk;
            #pragma unroll
            for (int jj = 0; jj < 8; ++jj) {
                float v = 0.0f;
                int dy = (rr * 11) >> 5;          // rr/3 for rr in [0,8]
                int dx = rr - dy * 3;
                if (cc < Ccat) {
                    int yy = y + dy - 1;
                    int xx = xg + dx - 1;
                    if ((unsigned)yy < (unsigned)H && (unsigned)xx < (unsigned)Wd) {
                        const float* base = (cc < Cin)
                            ? in  + (size_t)(b * Cin  + cc)        * HW
                            : hid + (size_t)(b * Chid + (cc - Cin)) * HW;
                        v = base[yy * Wd + xx];
                    }
                }
                pk.s[jj] = f2bf(v);
                rr += 1; if (rr == 9) { rr = 0; cc += 1; }
            }
            *(v4u*)(&sB[dflat]) = pk.u;
            rr += 6; cc += 2; if (rr >= 9) { rr -= 9; cc += 1; }   // +24 -> next block
        }
        wait_async0();
        __syncthreads();

        // ---- fragments: contiguous 32B per lane -> ds_load_b128 x2 each ----
        const int mrow  = wave * 16 + (lane & 15);
        const int khalf = lane >> 4;
        v16bf a  = load_frag(&sA[(mrow * 2 + khalf) * 16]);
        v16bf b0 = load_frag(&sB[(0 * 32 + lane) * 16]);
        v16bf b1 = load_frag(&sB[(1 * 32 + lane) * 16]);

        acc0 = __builtin_amdgcn_wmma_f32_16x16x32_bf16(false, a, false, b0,
                                                       (short)0, acc0, false, false);
        acc1 = __builtin_amdgcn_wmma_f32_16x16x32_bf16(false, a, false, b1,
                                                       (short)0, acc1, false, false);
        __syncthreads();
    }

    // C layout: VGPR r -> M = r (lanes 0-15) / 8+r (lanes 16-31), N = lane&15
    const int mbase = Mblk + wave * 16 + ((lane >> 4) ? 8 : 0);
    const int ng0   = Nbase + (lane & 15);
    const int ng1   = ng0 + 16;
    #pragma unroll
    for (int r = 0; r < 8; ++r) {
        int m = mbase + r;
        float bs = bias[m];
        gates[(size_t)m * Npos + ng0] = acc0[r] + bs;
        gates[(size_t)m * Npos + ng1] = acc1[r] + bs;
    }
}

// ---------------------------------------------------------------------------
// FC GEMM: out[n][m] = sum_k act[n][k] * Wp[m][k] + bias[m]
// Same tiling: WG = M64 x N32. act rows are contiguous K -> float4 staging.
// ---------------------------------------------------------------------------
__global__ __launch_bounds__(128)
void fc_gemm_wmma(const float* __restrict__ act,          // [N][K]
                  const unsigned short* __restrict__ Wp,  // [M][K] bf16
                  const float* __restrict__ bias,         // [M]
                  float* __restrict__ out,                // [N][M]
                  int Mdim, int K) {
    extern __shared__ unsigned short sMem[];
    unsigned short* sA = sMem;
    unsigned short* sB = sMem + LDS_A_HALVES;

    const int tid  = threadIdx.x;
    const int wave = tid >> 5;
    const int lane = tid & 31;
    const int Mblk  = blockIdx.y * 64;
    const int Nbase = blockIdx.x * 32;

    const int g0  = tid,        g1  = tid + 128;
    const int am0 = g0 >> 2,    am1 = g1 >> 2;
    const int aq0 = g0 & 3,     aq1 = g1 & 3;
    const unsigned adst0 = (unsigned)((am0 * 2 + (aq0 >> 1)) * 16 + (aq0 & 1) * 8) * 2;
    const unsigned adst1 = (unsigned)((am1 * 2 + (aq1 >> 1)) * 16 + (aq1 & 1) * 8) * 2;
    const unsigned short* asrc0 = Wp + (size_t)(Mblk + am0) * K + aq0 * 8;
    const unsigned short* asrc1 = Wp + (size_t)(Mblk + am1) * K + aq1 * 8;

    const int dflat = tid * 8;
    const int ti_st = dflat >> 9;
    const int remb  = dflat & 511;
    const int L_st  = remb >> 4;
    const int j0_st = remb & 15;
    const int nn_st = L_st & 15;
    const int kk0   = ((L_st >> 4) << 4) + j0_st;
    const int nrow  = Nbase + ti_st * 16 + nn_st;

    v8f acc0 = {}, acc1 = {};

    for (int k0 = 0; k0 < K; k0 += 32) {
        async_g2l_b128(adst0, asrc0 + k0);
        async_g2l_b128(adst1, asrc1 + k0);
        {
            const float* ap = act + (size_t)nrow * K + k0 + kk0;
            v4f f0 = *(const v4f*)(ap);
            v4f f1 = *(const v4f*)(ap + 4);
            Pack8 pk;
            #pragma unroll
            for (int jj = 0; jj < 4; ++jj) { pk.s[jj] = f2bf(f0[jj]); pk.s[jj + 4] = f2bf(f1[jj]); }
            *(v4u*)(&sB[dflat]) = pk.u;
        }
        wait_async0();
        __syncthreads();

        const int mrow  = wave * 16 + (lane & 15);
        const int khalf = lane >> 4;
        v16bf a  = load_frag(&sA[(mrow * 2 + khalf) * 16]);
        v16bf b0 = load_frag(&sB[(0 * 32 + lane) * 16]);
        v16bf b1 = load_frag(&sB[(1 * 32 + lane) * 16]);

        acc0 = __builtin_amdgcn_wmma_f32_16x16x32_bf16(false, a, false, b0,
                                                       (short)0, acc0, false, false);
        acc1 = __builtin_amdgcn_wmma_f32_16x16x32_bf16(false, a, false, b1,
                                                       (short)0, acc1, false, false);
        __syncthreads();
    }

    const int mbase = Mblk + wave * 16 + ((lane >> 4) ? 8 : 0);
    const int ng0   = Nbase + (lane & 15);
    const int ng1   = ng0 + 16;
    #pragma unroll
    for (int r = 0; r < 8; ++r) {
        int m = mbase + r;
        float bs = bias[m];
        out[(size_t)ng0 * Mdim + m] = acc0[r] + bs;
        out[(size_t)ng1 * Mdim + m] = acc1[r] + bs;
    }
}

// ---------------------------------------------------------------------------
// SConvLSTM state update: syn = sig(f)*syn + sig(i)*tanh(g); mem = sig(o)*tanh(syn)
// ---------------------------------------------------------------------------
__global__ void lstm_update(const float* __restrict__ gates,
                            float* __restrict__ syn, float* __restrict__ mem,
                            int Bn, int Ch, int HW) {
    int idx = blockIdx.x * 256 + threadIdx.x;
    int total = Bn * Ch * HW;
    if (idx >= total) return;
    int b = idx / (Ch * HW);
    int c = (idx / HW) % Ch;
    int p = idx % HW;
    size_t Npos = (size_t)Bn * HW;
    size_t n = (size_t)b * HW + p;
    float gi = gates[(size_t)(0 * Ch + c) * Npos + n];
    float gf = gates[(size_t)(1 * Ch + c) * Npos + n];
    float gg = gates[(size_t)(2 * Ch + c) * Npos + n];
    float go = gates[(size_t)(3 * Ch + c) * Npos + n];
    float s = sigf(gf) * syn[idx] + sigf(gi) * tanhf(gg);
    syn[idx] = s;
    mem[idx] = sigf(go) * tanhf(s);
}

// 2x2 maxpool on mem, heaviside spike
__global__ void pool_spike(const float* __restrict__ mem, float* __restrict__ spk,
                           int Bn, int Ch, int H, int Wd) {
    int Ho = H >> 1, Wo = Wd >> 1;
    int idx = blockIdx.x * 256 + threadIdx.x;
    int total = Bn * Ch * Ho * Wo;
    if (idx >= total) return;
    int xo = idx % Wo;
    int yo = (idx / Wo) % Ho;
    int c  = (idx / (Wo * Ho)) % Ch;
    int b  = idx / (Wo * Ho * Ch);
    const float* base = mem + (size_t)(b * Ch + c) * H * Wd;
    float m0 = base[(2 * yo) * Wd + 2 * xo];
    float m1 = base[(2 * yo) * Wd + 2 * xo + 1];
    float m2 = base[(2 * yo + 1) * Wd + 2 * xo];
    float m3 = base[(2 * yo + 1) * Wd + 2 * xo + 1];
    float mx = fmaxf(fmaxf(m0, m1), fmaxf(m2, m3));
    spk[idx] = ((mx - THR) > 0.0f) ? 1.0f : 0.0f;
}

// Leaky (subtract reset from previous mem, detached): fwd heaviside spike
__global__ void leaky_update(const float* __restrict__ cur, float* __restrict__ mem,
                             float* __restrict__ spk, int total) {
    int idx = blockIdx.x * 256 + threadIdx.x;
    if (idx >= total) return;
    float m = mem[idx];
    float reset = (m > THR) ? 1.0f : 0.0f;
    m = BETA * m + cur[idx] - reset * THR;
    mem[idx] = m;
    spk[idx] = ((m - THR) > 0.0f) ? 1.0f : 0.0f;
}

// FC2 (64x2, K=512) + leaky + write outputs for step t. 128 threads: (b,o).
__global__ void fc2_leaky_out(const float* __restrict__ spk4,
                              const float* __restrict__ w,    // [2][512]
                              const float* __restrict__ bias, // [2]
                              float* __restrict__ mem5,       // [128]
                              float* __restrict__ out, int t) {
    int tid = threadIdx.x;
    int b = tid >> 1, o = tid & 1;
    float dot = bias[o];
    for (int k = 0; k < 512; ++k)
        dot += spk4[b * 512 + k] * w[o * 512 + k];
    float m = mem5[tid];
    float reset = (m > THR) ? 1.0f : 0.0f;
    m = BETA * m + dot - reset * THR;
    mem5[tid] = m;
    out[t * 128 + tid]        = ((m - THR) > 0.0f) ? 1.0f : 0.0f;  // spk_rec
    out[3200 + t * 128 + tid] = m;                                 // mem_rec
}

// ---------------------------------------------------------------------------
extern "C" void kernel_launch(void* const* d_in, const int* in_sizes, int n_in,
                              void* d_out, int out_size, void* d_ws, size_t ws_size,
                              hipStream_t stream) {
    (void)in_sizes; (void)n_in; (void)out_size; (void)ws_size;

    const float* x     = (const float*)d_in[0];   // [25][64][16][128]
    const float* w1    = (const float*)d_in[1];   // [64][17][3][3]
    const float* b1    = (const float*)d_in[2];
    const float* w2    = (const float*)d_in[3];   // [128][48][3][3]
    const float* b2    = (const float*)d_in[4];
    const float* w3    = (const float*)d_in[5];   // [256][96][3][3]
    const float* b3    = (const float*)d_in[6];
    const float* fc1w  = (const float*)d_in[7];   // [512][2048]
    const float* fc1b  = (const float*)d_in[8];
    const float* fc2w  = (const float*)d_in[9];   // [2][512]
    const float* fc2b  = (const float*)d_in[10];
    float* out = (float*)d_out;

    const int B = 64;

    // ---- workspace carve-up (floats first, then bf16 weight packs) ----
    char* ws = (char*)d_ws;
    size_t off = 0;
    auto allocf = [&](size_t n) { float* p = (float*)(ws + off); off += n * sizeof(float); return p; };

    float* syn1 = allocf((size_t)B * 16 * 16 * 128);
    float* mem1 = allocf((size_t)B * 16 * 16 * 128);
    float* spk1 = allocf((size_t)B * 16 * 8 * 64);
    float* syn2 = allocf((size_t)B * 32 * 8 * 64);
    float* mem2 = allocf((size_t)B * 32 * 8 * 64);
    float* spk2 = allocf((size_t)B * 32 * 4 * 32);
    float* syn3 = allocf((size_t)B * 64 * 4 * 32);
    float* mem3 = allocf((size_t)B * 64 * 4 * 32);
    float* spk3 = allocf((size_t)B * 2048);
    float* mem4 = allocf((size_t)B * 512);
    float* spk4 = allocf((size_t)B * 512);
    float* cur4 = allocf((size_t)B * 512);
    float* mem5 = allocf(128);
    size_t state_bytes = off;                          // zero all state each call

    float* gates = allocf((size_t)64 * B * 16 * 128);  // shared gate buffer (max layer)

    auto allocb = [&](size_t n) { unsigned short* p = (unsigned short*)(ws + off); off += n * 2; return p; };
    unsigned short* Wp1 = allocb((size_t)64  * 160);
    unsigned short* Wp2 = allocb((size_t)128 * 448);
    unsigned short* Wp3 = allocb((size_t)256 * 864);
    unsigned short* Wfc = allocb((size_t)512 * 2048);

    hipMemsetAsync(d_ws, 0, state_bytes, stream);

    // ---- weight repack (once per launch; deterministic) ----
    auto rep = [&](const float* src, unsigned short* dst, int Cout, int Kt, int Kp) {
        int total = Cout * Kp;
        repack_bf16<<<(total + 255) / 256, 256, 0, stream>>>(src, dst, Cout, Kt, Kp);
    };
    rep(w1,   Wp1, 64,  153,  160);
    rep(w2,   Wp2, 128, 432,  448);
    rep(w3,   Wp3, 256, 864,  864);
    rep(fc1w, Wfc, 512, 2048, 2048);

    // ---- time loop ----
    const int N1 = B * 16 * 128;   // 131072 positions
    const int N2 = B * 8 * 64;     //  32768
    const int N3 = B * 4 * 32;     //   8192

    for (int t = 0; t < 25; ++t) {
        const float* xt = x + (size_t)t * B * 16 * 128;

        // layer 1: Cin=1, Chid=16, 16x128
        conv_gates_wmma<<<dim3(N1 / 32, 1), 128, LDS_BYTES, stream>>>(
            xt, mem1, Wp1, b1, gates, B, 1, 16, 16, 128, 160);
        lstm_update<<<(B * 16 * 2048 + 255) / 256, 256, 0, stream>>>(
            gates, syn1, mem1, B, 16, 16 * 128);
        pool_spike<<<(B * 16 * 8 * 64 + 255) / 256, 256, 0, stream>>>(
            mem1, spk1, B, 16, 16, 128);

        // layer 2: Cin=16, Chid=32, 8x64
        conv_gates_wmma<<<dim3(N2 / 32, 2), 128, LDS_BYTES, stream>>>(
            spk1, mem2, Wp2, b2, gates, B, 16, 32, 8, 64, 448);
        lstm_update<<<(B * 32 * 512 + 255) / 256, 256, 0, stream>>>(
            gates, syn2, mem2, B, 32, 8 * 64);
        pool_spike<<<(B * 32 * 4 * 32 + 255) / 256, 256, 0, stream>>>(
            mem2, spk2, B, 32, 8, 64);

        // layer 3: Cin=32, Chid=64, 4x32
        conv_gates_wmma<<<dim3(N3 / 32, 4), 128, LDS_BYTES, stream>>>(
            spk2, mem3, Wp3, b3, gates, B, 32, 64, 4, 32, 864);
        lstm_update<<<(B * 64 * 128 + 255) / 256, 256, 0, stream>>>(
            gates, syn3, mem3, B, 64, 4 * 32);
        pool_spike<<<(B * 64 * 2 * 16 + 255) / 256, 256, 0, stream>>>(
            mem3, spk3, B, 64, 4, 32);          // spk3 flat == reshape(B, 2048)

        // fc1 (512x2048) + leaky
        fc_gemm_wmma<<<dim3(B / 32, 512 / 64), 128, LDS_BYTES, stream>>>(
            spk3, Wfc, fc1b, cur4, 512, 2048);
        leaky_update<<<(B * 512 + 255) / 256, 256, 0, stream>>>(
            cur4, mem4, spk4, B * 512);

        // fc2 (2x512) + leaky + record
        fc2_leaky_out<<<1, 128, 0, stream>>>(spk4, fc2w, fc2b, mem5, out, t);
    }
}
